// Seq2SeqGAN_87041807221333
// MI455X (gfx1250) — matrix operand order repaired and checked by reading, not compiled, not run on hardware
//
#include <hip/hip_runtime.h>
#include <hip/hip_bf16.h>
#include <math.h>

// ---------------------------------------------------------------------------
// Seq2Seq LSTM (2-layer encoder 168 steps, 2-layer AR decoder 96 steps)
// B=512, H=512, gates=2048. Mixed precision: bf16 WMMA GEMM, f32 state.
// Fused [x_t | h_{t-1}] x [W_ih | W_hh]^T GEMM per layer per step.
// Weight tiles staged LDS via CDNA5 async copy (GLOBAL_LOAD_ASYNC_TO_LDS).
// ---------------------------------------------------------------------------

#define Bsz   512
#define Hdim  512
#define G4H   2048
#define T_H   168
#define T_F   96

typedef __attribute__((ext_vector_type(16))) __bf16 v16bf;
typedef __attribute__((ext_vector_type(8)))  float  v8f;
typedef unsigned int u32;
typedef __attribute__((ext_vector_type(4)))  u32    v4u;

union FragU { v16bf bf; v4u u4[2]; };

static __device__ __forceinline__ unsigned short f2bf(float x) {
    u32 u = __float_as_uint(x);
    u32 r = (u + 0x7FFFu + ((u >> 16) & 1u)) >> 16;   // round-to-nearest-even
    return (unsigned short)r;
}
static __device__ __forceinline__ float sigf(float x) {
    return 1.0f / (1.0f + __expf(-x));
}

// LDS byte-offset of a __shared__ object (ptrtoint of addrspace(3) pointer)
static __device__ __forceinline__ unsigned lds_offset(const void* p) {
    return (unsigned)(unsigned long long)
        (const __attribute__((address_space(3))) void*)(p);
}

// ---------------------------------------------------------------------------
// Pack [W_ih (4H x kin) | zero-pad to kinpad | W_hh (4H x 512)] -> bf16 [4H][ktot]
// ---------------------------------------------------------------------------
__global__ void pack_w(unsigned short* __restrict__ dst,
                       const float* __restrict__ wih,
                       const float* __restrict__ whh,
                       int kin, int kinpad, int ktot) {
    int total = G4H * ktot;
    for (int i = blockIdx.x * blockDim.x + threadIdx.x; i < total;
         i += gridDim.x * blockDim.x) {
        int n = i / ktot, k = i - n * ktot;
        float v = (k < kin) ? wih[n * kin + k]
                : (k < kinpad) ? 0.0f
                : whh[n * Hdim + (k - kinpad)];
        dst[i] = f2bf(v);
    }
}

__global__ void zero_f(float* __restrict__ p, int n) {
    for (int i = blockIdx.x * blockDim.x + threadIdx.x; i < n;
         i += gridDim.x * blockDim.x) p[i] = 0.0f;
}

// ---------------------------------------------------------------------------
// One LSTM step for one layer: gates = [x_t | h] @ W^T + b, then cell update.
// Grid: (16, 4)  -> blockIdx.x selects 32 hidden units, blockIdx.y 128 rows.
// Block: 256 threads = 8 wave32; wave w owns rows [w*16, w*16+16).
// Each wave accumulates 4 gates x 2 column-tiles = 8 wmma 16x16 tiles.
// mode 0: encoder L0 (x_hist(8)+emb(16), K=544)
// mode 1: layer 1      (h_low(512)+h(512), K=1024)
// mode 2: decoder L0 (y_prev(1)+exo(4)+emb(16) + z_bias, K=544)
// ---------------------------------------------------------------------------
__global__ __launch_bounds__(256)
void lstm_step(const unsigned short* __restrict__ W,   // [2048][Ktot] bf16
               const float* __restrict__ bias,          // [2048]
               const float* __restrict__ hin,           // [512][512]
               float* __restrict__ hout,                // [512][512]
               float* __restrict__ cbuf,                // [512][512] in-place
               int Ktot, int mode, int t,
               const float* __restrict__ srcA,          // x_hist / h_low / x_future
               const float* __restrict__ emb,           // [200][16]
               const int*   __restrict__ turb,          // [512]
               const float* __restrict__ zb,            // [21] (zproj_b)
               const float* __restrict__ yprev, int ystride) {
    __shared__ __align__(16) unsigned short Ash[128][32];  // A tile  (rows x K32)
    __shared__ __align__(16) unsigned short Bsh[128][32];  // W tile  (4 gates x32 x K32)

    const int tid  = threadIdx.x;
    const int wv   = tid >> 5;
    const int lane = tid & 31;
    const int lrow = lane & 15;
    const int lhi  = lane >> 4;
    const int mTile = blockIdx.y * 128;        // batch-row base
    const int nTile = blockIdx.x * 32;         // hidden-unit base

    v8f acc[4][2];
    #pragma unroll
    for (int g = 0; g < 4; ++g)
        #pragma unroll
        for (int j = 0; j < 2; ++j) acc[g][j] = (v8f){0,0,0,0,0,0,0,0};

    // staging assignments
    const int a_ml    = tid >> 1;              // A row this thread stages
    const int a_kpart = (tid & 1) * 16;        // A k sub-range

    // B tile: 128 rows x 32 k of bf16 = 512 chunks of 16B; 2 async copies/thread
    const int  b_nl0 = tid >> 2;                       // chunk set 0: rows 0..63
    const int  b_nl1 = (tid + 256) >> 2;               // chunk set 1: rows 64..127
    const int  b_part = tid & 3;
    const int  b_ng0 = (b_nl0 >> 5) * Hdim + nTile + (b_nl0 & 31);
    const int  b_ng1 = (b_nl1 >> 5) * Hdim + nTile + (b_nl1 & 31);
    const unsigned b_lds0 = lds_offset(&Bsh[b_nl0][b_part * 8]);
    const unsigned b_lds1 = lds_offset(&Bsh[b_nl1][b_part * 8]);
    const unsigned long long wbase = (unsigned long long)W;

    for (int k0 = 0; k0 < Ktot; k0 += 32) {
        // ---- async-copy W tile (row-major [n][k]) directly into LDS --------
        {
            const unsigned go0 = (unsigned)(((size_t)b_ng0 * Ktot + k0 + b_part * 8) * 2);
            const unsigned go1 = (unsigned)(((size_t)b_ng1 * Ktot + k0 + b_part * 8) * 2);
            asm volatile("global_load_async_to_lds_b128 %0, %1, %2"
                         :: "v"(b_lds0), "v"(go0), "s"(wbase) : "memory");
            asm volatile("global_load_async_to_lds_b128 %0, %1, %2"
                         :: "v"(b_lds1), "v"(go1), "s"(wbase) : "memory");
        }
        // ---- stage A = [x_t | h_{t-1}] -> bf16 LDS (needs f32->bf16) -------
        {
            const int m = mTile + a_ml;
            #pragma unroll
            for (int i = 0; i < 16; ++i) {
                const int k = k0 + a_kpart + i;
                float v;
                if (mode == 1) {
                    v = (k < Hdim) ? srcA[(size_t)m * Hdim + k]
                                   : hin[(size_t)m * Hdim + (k - Hdim)];
                } else if (k >= 32) {
                    v = hin[(size_t)m * Hdim + (k - 32)];
                } else if (mode == 0) {
                    if (k < 8)       v = srcA[((size_t)m * T_H + t) * 8 + k];
                    else if (k < 24) v = emb[(size_t)turb[m] * 16 + (k - 8)];
                    else             v = 0.0f;
                } else { // mode 2
                    if (k == 0)      v = yprev[(size_t)m * ystride] + zb[0];
                    else if (k < 5)  v = srcA[((size_t)m * T_F + t) * 4 + (k - 1)] + zb[k];
                    else if (k < 21) v = emb[(size_t)turb[m] * 16 + (k - 5)] + zb[k];
                    else             v = 0.0f;
                }
                Ash[a_ml][a_kpart + i] = f2bf(v);
            }
        }
        // all async copies of this wave done, then block-wide visibility
        asm volatile("s_wait_asynccnt 0" ::: "memory");
        __syncthreads();

        // ---- A fragment (16x32 bf16, per ISA VGPR layout) ------------------
        FragU fa;
        fa.u4[0] = *reinterpret_cast<const v4u*>(&Ash[wv * 16 + lrow][lhi * 8]);
        fa.u4[1] = *reinterpret_cast<const v4u*>(&Ash[wv * 16 + lrow][16 + lhi * 8]);

        // ---- 4 gates x 2 tiles: B fragment + wmma --------------------------
        #pragma unroll
        for (int g = 0; g < 4; ++g) {
            #pragma unroll
            for (int j = 0; j < 2; ++j) {
                FragU fb;
                fb.u4[0] = *reinterpret_cast<const v4u*>(
                    &Bsh[g * 32 + j * 16 + lrow][lhi * 8]);
                fb.u4[1] = *reinterpret_cast<const v4u*>(
                    &Bsh[g * 32 + j * 16 + lrow][16 + lhi * 8]);
                acc[g][j] = __builtin_amdgcn_wmma_f32_16x16x32_bf16(
                    false, fa.bf, false, fb.bf, (short)0, acc[g][j], false, false);
            }
        }
        __syncthreads();
    }

    // ---- LSTM elementwise: all 4 gates local to this lane ------------------
    #pragma unroll
    for (int j = 0; j < 2; ++j) {
        const int hu = nTile + j * 16 + lrow;        // hidden unit
        const float bi = bias[hu];
        const float bf = bias[Hdim + hu];
        const float bg = bias[2 * Hdim + hu];
        const float bo = bias[3 * Hdim + hu];
        #pragma unroll
        for (int r = 0; r < 8; ++r) {
            const int m = mTile + wv * 16 + r + 8 * lhi; // C-layout: M = r + 8*(lane>=16)
            const size_t idx = (size_t)m * Hdim + hu;
            const float iv = acc[0][j][r] + bi;
            const float fv = acc[1][j][r] + bf;
            const float gv = acc[2][j][r] + bg;
            const float ov = acc[3][j][r] + bo;
            const float cn = sigf(fv) * cbuf[idx] + sigf(iv) * tanhf(gv);
            const float hn = sigf(ov) * tanhf(cn);
            cbuf[idx] = cn;
            hout[idx] = hn;
        }
    }
}

// ---------------------------------------------------------------------------
// yhat = h1 @ proj_W^T + proj_b  ->  out[:, t]
// ---------------------------------------------------------------------------
__global__ __launch_bounds__(256)
void proj_kernel(const float* __restrict__ h1, const float* __restrict__ pW,
                 const float* __restrict__ pb, float* __restrict__ out, int t) {
    const int wv = threadIdx.x >> 5, lane = threadIdx.x & 31;
    const int m = blockIdx.x * 8 + wv;
    float s = 0.0f;
    #pragma unroll
    for (int k = lane; k < Hdim; k += 32) s += h1[(size_t)m * Hdim + k] * pW[k];
    #pragma unroll
    for (int off = 16; off; off >>= 1) s += __shfl_xor(s, off, 32);
    if (lane == 0) out[(size_t)m * T_F + t] = s + pb[0];
}

// ---------------------------------------------------------------------------
extern "C" void kernel_launch(void* const* d_in, const int* in_sizes, int n_in,
                              void* d_out, int out_size, void* d_ws, size_t ws_size,
                              hipStream_t stream) {
    const float* x_hist   = (const float*)d_in[0];
    const float* x_future = (const float*)d_in[1];
    const float* y0       = (const float*)d_in[2];
    const int*   turb     = (const int*)  d_in[3];
    const float* emb      = (const float*)d_in[5];
    const float* eW_ih0   = (const float*)d_in[6];
    const float* eW_hh0   = (const float*)d_in[7];
    const float* eb0      = (const float*)d_in[8];
    const float* eW_ih1   = (const float*)d_in[9];
    const float* eW_hh1   = (const float*)d_in[10];
    const float* eb1      = (const float*)d_in[11];
    const float* dW_ih0   = (const float*)d_in[12];
    const float* dW_hh0   = (const float*)d_in[13];
    const float* db0      = (const float*)d_in[14];
    const float* dW_ih1   = (const float*)d_in[15];
    const float* dW_hh1   = (const float*)d_in[16];
    const float* db1      = (const float*)d_in[17];
    const float* projW    = (const float*)d_in[18];
    const float* projb    = (const float*)d_in[19];
    const float* zprojb   = (const float*)d_in[21];
    float* out = (float*)d_out;

    // workspace carve-up
    size_t off = 0;
    auto take = [&](size_t bytes) -> char* {
        char* p = (char*)d_ws + off;
        off += (bytes + 255) & ~(size_t)255;
        return p;
    };
    const int K0 = 544, K1 = 1024;
    unsigned short* We0 = (unsigned short*)take((size_t)G4H * K0 * 2);
    unsigned short* We1 = (unsigned short*)take((size_t)G4H * K1 * 2);
    unsigned short* Wd0 = (unsigned short*)take((size_t)G4H * K0 * 2);
    unsigned short* Wd1 = (unsigned short*)take((size_t)G4H * K1 * 2);
    const size_t SN = (size_t)Bsz * Hdim;                  // 262144 floats
    float* states = (float*)take(6 * SN * sizeof(float));
    float* h0buf[2] = { states + 0 * SN, states + 1 * SN };
    float* c0buf    =   states + 2 * SN;
    float* h1buf[2] = { states + 3 * SN, states + 4 * SN };
    float* c1buf    =   states + 5 * SN;

    // one-time prep (runs every launch: deterministic)
    pack_w<<<2048, 256, 0, stream>>>(We0, eW_ih0, eW_hh0, 24, 32, K0);
    pack_w<<<2048, 256, 0, stream>>>(We1, eW_ih1, eW_hh1, Hdim, Hdim, K1);
    pack_w<<<2048, 256, 0, stream>>>(Wd0, dW_ih0, dW_hh0, 21, 32, K0);
    pack_w<<<2048, 256, 0, stream>>>(Wd1, dW_ih1, dW_hh1, Hdim, Hdim, K1);
    zero_f<<<2048, 256, 0, stream>>>(states, (int)(6 * SN));

    dim3 grid(16, 4);

    // ---- encoder: layer0 / layer1 interleaved per step (exact) -------------
    for (int t = 0; t < T_H; ++t) {
        const int p = t & 1;
        lstm_step<<<grid, 256, 0, stream>>>(We0, eb0, h0buf[p], h0buf[1 - p], c0buf,
                                            K0, 0, t, x_hist, emb, turb,
                                            nullptr, nullptr, 0);
        lstm_step<<<grid, 256, 0, stream>>>(We1, eb1, h1buf[p], h1buf[1 - p], c1buf,
                                            K1, 1, t, h0buf[1 - p], emb, turb,
                                            nullptr, nullptr, 0);
    }

    // ---- decoder: autoregressive, yhat fed back through d_out --------------
    for (int t = 0; t < T_F; ++t) {
        const int s = T_H + t;
        const int p = s & 1;
        const float* yp = (t == 0) ? y0 : (out + (t - 1));
        const int ystride = (t == 0) ? 1 : T_F;
        lstm_step<<<grid, 256, 0, stream>>>(Wd0, db0, h0buf[p], h0buf[1 - p], c0buf,
                                            K0, 2, t, x_future, emb, turb,
                                            zprojb, yp, ystride);
        lstm_step<<<grid, 256, 0, stream>>>(Wd1, db1, h1buf[p], h1buf[1 - p], c1buf,
                                            K1, 1, t, h0buf[1 - p], emb, turb,
                                            nullptr, nullptr, 0);
        proj_kernel<<<64, 256, 0, stream>>>(h1buf[1 - p], projW, projb, out, t);
    }
}
